// VoxelProposal_52158082843044
// MI455X (gfx1250) — compile-verified, matrix operand order checked by reading, not run on hardware
//
#include <hip/hip_runtime.h>

// ---------------------------------------------------------------------------
// Sizes from the reference
// ---------------------------------------------------------------------------
#define EMBED 128
#define HEADS 8
#define DH 16
#define NVOXT 131072            // 128*128*8
#define NVF 9554                // 47*154 + 24*77 + 12*39
#define MT_VAL 598              // ceil(NVF/16)
#define IMGH 370
#define IMGW 1226
static __device__ __constant__ int c_lh[3] = {47, 24, 12};
static __device__ __constant__ int c_lw[3] = {154, 77, 39};
static __device__ __constant__ int c_ls[3] = {0, 7238, 9086};
#define S1 7238
#define S2 9086
#define HW0 7238
#define HW1 1848
#define HW2 468

typedef __attribute__((ext_vector_type(16))) _Float16 v16h;
typedef __attribute__((ext_vector_type(8)))  _Float16 v8h;
typedef __attribute__((ext_vector_type(8)))  float    v8f;

// ---------------------------------------------------------------------------
// WMMA layout helpers (CDNA5 v_wmma_f32_16x16x32_f16, wave32)
//  A: lane L holds row m=L&15; half-slot i -> k32 = (i<8 ? i : i+8) + 8*(L>=16)
//     => slots 0..7 are k = kb..kb+7 and slots 8..15 are k = kb+16..kb+23,
//        kb = 8*(L>=16): two CONTIGUOUS 8-half runs from a row-major tile.
//  B mirrors A with column n = L&15 (packed in ws as 32B/lane contiguous).
//  D: VGPR v -> m = v + 8*(L>=16), n = L&15.
// ---------------------------------------------------------------------------
__device__ __forceinline__ int kmap(int lane, int i) {
    int base = (i < 8) ? i : (i + 8);
    return base + ((lane & 16) ? 8 : 0);
}

// A fragment from a row-major f16 LDS tile: 2x ds_load_b128
__device__ __forceinline__ v16h ldsA(const _Float16* s, int lane, int stride, int k0) {
    int m = lane & 15;
    int kb = (lane & 16) ? 8 : 0;
    const v8h* p = (const v8h*)(s + m * stride + k0 + kb);
    v8h lo = p[0];
    v8h hi = p[2];                       // +16 halfs
    return __builtin_shufflevector(lo, hi, 0, 1, 2, 3, 4, 5, 6, 7,
                                   8, 9, 10, 11, 12, 13, 14, 15);
}

// B fragment from a packed weight: 32 contiguous bytes/lane (2x global_load_b128)
__device__ __forceinline__ v16h pkB(const _Float16* __restrict__ P, int KT,
                                    int kt, int nt, int lane) {
    return *(const v16h*)(P + (((size_t)(nt * KT + kt)) * 32 + lane) * 16);
}

__device__ __forceinline__ v8f wmma_f16(v16h a, v16h b, v8f c) {
    return __builtin_amdgcn_wmma_f32_16x16x32_f16(false, a, false, b, (short)0, c,
                                                  false, false);
}

// ---------------------------------------------------------------------------
// Kernel 0: 3x3 inverse of K and 4x4 inverse of E -> ws[0..24]
// ---------------------------------------------------------------------------
__global__ void k_setup(const float* __restrict__ K, const float* __restrict__ E,
                        float* __restrict__ invKE) {
    if (threadIdx.x != 0 || blockIdx.x != 0) return;
    {
        float a = K[0], b = K[1], c = K[2], d = K[3], e = K[4], f = K[5],
              g = K[6], h = K[7], i = K[8];
        float A = e * i - f * h, B = -(d * i - f * g), C = d * h - e * g;
        float rd = 1.0f / (a * A + b * B + c * C);
        invKE[0] = A * rd; invKE[1] = (c * h - b * i) * rd; invKE[2] = (b * f - c * e) * rd;
        invKE[3] = B * rd; invKE[4] = (a * i - c * g) * rd; invKE[5] = (c * d - a * f) * rd;
        invKE[6] = C * rd; invKE[7] = (b * g - a * h) * rd; invKE[8] = (a * e - b * d) * rd;
    }
    {
        float m[16], inv[16];
        for (int j = 0; j < 16; ++j) m[j] = E[j];
        inv[0]  =  m[5]*m[10]*m[15] - m[5]*m[11]*m[14] - m[9]*m[6]*m[15] + m[9]*m[7]*m[14] + m[13]*m[6]*m[11] - m[13]*m[7]*m[10];
        inv[4]  = -m[4]*m[10]*m[15] + m[4]*m[11]*m[14] + m[8]*m[6]*m[15] - m[8]*m[7]*m[14] - m[12]*m[6]*m[11] + m[12]*m[7]*m[10];
        inv[8]  =  m[4]*m[9]*m[15]  - m[4]*m[11]*m[13] - m[8]*m[5]*m[15] + m[8]*m[7]*m[13] + m[12]*m[5]*m[11] - m[12]*m[7]*m[9];
        inv[12] = -m[4]*m[9]*m[14]  + m[4]*m[10]*m[13] + m[8]*m[5]*m[14] - m[8]*m[6]*m[13] - m[12]*m[5]*m[10] + m[12]*m[6]*m[9];
        inv[1]  = -m[1]*m[10]*m[15] + m[1]*m[11]*m[14] + m[9]*m[2]*m[15] - m[9]*m[3]*m[14] - m[13]*m[2]*m[11] + m[13]*m[3]*m[10];
        inv[5]  =  m[0]*m[10]*m[15] - m[0]*m[11]*m[14] - m[8]*m[2]*m[15] + m[8]*m[3]*m[14] + m[12]*m[2]*m[11] - m[12]*m[3]*m[10];
        inv[9]  = -m[0]*m[9]*m[15]  + m[0]*m[11]*m[13] + m[8]*m[1]*m[15] - m[8]*m[3]*m[13] - m[12]*m[1]*m[11] + m[12]*m[3]*m[9];
        inv[13] =  m[0]*m[9]*m[14]  - m[0]*m[10]*m[13] - m[8]*m[1]*m[14] + m[8]*m[2]*m[13] + m[12]*m[1]*m[10] - m[12]*m[2]*m[9];
        inv[2]  =  m[1]*m[6]*m[15]  - m[1]*m[7]*m[14]  - m[5]*m[2]*m[15] + m[5]*m[3]*m[14] + m[13]*m[2]*m[7]  - m[13]*m[3]*m[6];
        inv[6]  = -m[0]*m[6]*m[15]  + m[0]*m[7]*m[14]  + m[4]*m[2]*m[15] - m[4]*m[3]*m[14] - m[12]*m[2]*m[7]  + m[12]*m[3]*m[6];
        inv[10] =  m[0]*m[5]*m[15]  - m[0]*m[7]*m[13]  - m[4]*m[1]*m[15] + m[4]*m[3]*m[13] + m[12]*m[1]*m[7]  - m[12]*m[3]*m[5];
        inv[14] = -m[0]*m[5]*m[14]  + m[0]*m[6]*m[13]  + m[4]*m[1]*m[14] - m[4]*m[2]*m[13] - m[12]*m[1]*m[6]  + m[12]*m[2]*m[5];
        inv[3]  = -m[1]*m[6]*m[11]  + m[1]*m[7]*m[10]  + m[5]*m[2]*m[11] - m[5]*m[3]*m[10] - m[9]*m[2]*m[7]   + m[9]*m[3]*m[6];
        inv[7]  =  m[0]*m[6]*m[11]  - m[0]*m[7]*m[10]  - m[4]*m[2]*m[11] + m[4]*m[3]*m[10] + m[8]*m[2]*m[7]   - m[8]*m[3]*m[6];
        inv[11] = -m[0]*m[5]*m[11]  + m[0]*m[7]*m[9]   + m[4]*m[1]*m[11] - m[4]*m[3]*m[9]  - m[8]*m[1]*m[7]   + m[8]*m[3]*m[5];
        inv[15] =  m[0]*m[5]*m[10]  - m[0]*m[6]*m[9]   - m[4]*m[1]*m[10] + m[4]*m[2]*m[9]  + m[8]*m[1]*m[6]   - m[8]*m[2]*m[5];
        float r4 = 1.0f / (m[0]*inv[0] + m[1]*inv[4] + m[2]*inv[8] + m[3]*inv[12]);
        for (int j = 0; j < 16; ++j) invKE[9 + j] = inv[j] * r4;
    }
}

__global__ void k_zero(int* __restrict__ mask) {
    int i = blockIdx.x * 256 + threadIdx.x;
    if (i < NVOXT) mask[i] = 0;
}

// ---------------------------------------------------------------------------
// Kernel 1: depth back-projection -> voxel occupancy mask
// ---------------------------------------------------------------------------
__global__ void k_project(const float* __restrict__ depth,
                          const float* __restrict__ invKE,
                          const float* __restrict__ origin,
                          int* __restrict__ mask) {
    int i = blockIdx.x * 256 + threadIdx.x;
    if (i >= IMGH * IMGW) return;
    int v = i / IMGW, u = i - v * IMGW;
    float d = depth[i];
    float px = (float)u * d, py = (float)v * d, pz = d;
    const float* iK = invKE;
    const float* iE = invKE + 9;
    float cx = iK[0] * px + iK[1] * py + iK[2] * pz;
    float cy = iK[3] * px + iK[4] * py + iK[5] * pz;
    float cz = iK[6] * px + iK[7] * py + iK[8] * pz;
    float wx = iE[0] * cx + iE[1] * cy + iE[2]  * cz + iE[3];
    float wy = iE[4] * cx + iE[5] * cy + iE[6]  * cz + iE[7];
    float wz = iE[8] * cx + iE[9] * cy + iE[10] * cz + iE[11];
    int vx = (int)truncf((wx - origin[0]) * 2.5f - 0.5f);   // 1/VOXEL_SIZE
    int vy = (int)truncf((wy - origin[1]) * 2.5f - 0.5f);
    int vz = (int)truncf((wz - origin[2]) * 2.5f - 0.5f);
    if (vx >= 0 && vx < 128 && vy >= 0 && vy < 128 && vz >= 0 && vz < 8)
        atomicOr(&mask[(vx * 128 + vy) * 8 + vz], 1);
}

// ---------------------------------------------------------------------------
// Kernel 2a: pack a row-major f32 weight W[K x N] into WMMA-B f16 layout.
// ---------------------------------------------------------------------------
__global__ void k_packW(const float* __restrict__ W, int KT, int N,
                        _Float16* __restrict__ P) {
    int t = blockIdx.x * 256 + threadIdx.x;
    int total = KT * (N >> 4) * 32;
    if (t >= total) return;
    int lane = t & 31, tile = t >> 5;
    int kt = tile % KT, nt = tile / KT;
    int n = nt * 16 + (lane & 15);
    alignas(32) _Float16 tmp[16];
#pragma unroll
    for (int i = 0; i < 16; ++i) {
        int k = kt * 32 + kmap(lane, i);
        tmp[i] = (_Float16)W[(size_t)k * N + n];
    }
    *(v16h*)(P + (size_t)t * 16) = *(const v16h*)tmp;
}

// ---------------------------------------------------------------------------
// Kernel 2b: pack feat_flat (NVF x 128) into WMMA-A f16 layout (rows clamped).
// ---------------------------------------------------------------------------
__global__ void k_packA(const float* __restrict__ f0, const float* __restrict__ f1,
                        const float* __restrict__ f2, _Float16* __restrict__ PA) {
    int t = blockIdx.x * 256 + threadIdx.x;
    int total = MT_VAL * 4 * 32;
    if (t >= total) return;
    int lane = t & 31, tile = t >> 5;
    int kt = tile & 3, mt = tile >> 2;
    int row = mt * 16 + (lane & 15);
    int rowc = row < NVF ? row : NVF - 1;
    const float* base = f0; int hw = HW0, pos = rowc;
    if (rowc >= S2)      { base = f2; hw = HW2; pos = rowc - S2; }
    else if (rowc >= S1) { base = f1; hw = HW1; pos = rowc - S1; }
    alignas(32) _Float16 tmp[16];
#pragma unroll
    for (int i = 0; i < 16; ++i) {
        int k = kt * 32 + kmap(lane, i);
        tmp[i] = (_Float16)base[(size_t)k * hw + pos];
    }
    *(v16h*)(PA + (size_t)t * 16) = *(const v16h*)tmp;
}

// ---------------------------------------------------------------------------
// Kernel 2c: value = feat_flat @ W_val + b_val  (pure b128 loads + WMMA)
// ---------------------------------------------------------------------------
__global__ __launch_bounds__(256, 1)
void k_value(const _Float16* __restrict__ PA, const _Float16* __restrict__ pVal,
             const float* __restrict__ bv, float* __restrict__ val) {
    int tid = threadIdx.x, wave = tid >> 5, lane = tid & 31;
    int mt = blockIdx.x;
    v8f c = {};
#pragma unroll
    for (int ks = 0; ks < 4; ++ks) {
        v16h a = *(const v16h*)(PA + (((size_t)(mt * 4 + ks)) * 32 + lane) * 16);
        v16h b = pkB(pVal, 4, ks, wave, lane);
        c = wmma_f16(a, b, c);
    }
    int n = lane & 15, mo = (lane & 16) ? 8 : 0;
    int col = wave * 16 + n;
    float bb = bv[col];                      // uniform across the 8 D rows
    int rbase = mt * 16;
    if (rbase + 16 <= NVF) {                 // interior tile: unguarded stores
#pragma unroll
        for (int v = 0; v < 8; ++v)
            val[(size_t)(rbase + v + mo) * EMBED + col] = c[v] + bb;
    } else {                                 // only the last tile (1 of 598)
#pragma unroll
        for (int v = 0; v < 8; ++v) {
            int rr = rbase + v + mo;
            if (rr < NVF) val[(size_t)rr * EMBED + col] = c[v] + bb;
        }
    }
}

// ---------------------------------------------------------------------------
// Kernel 3: fused per-16-voxel mega-kernel (8 waves).
// ---------------------------------------------------------------------------
__global__ __launch_bounds__(256, 1)
void k_mega(const float* __restrict__ qg, const float* __restrict__ refpix,
            const float* __restrict__ val, const int* __restrict__ mask,
            const _Float16* __restrict__ pOff, const float* __restrict__ boff,
            const _Float16* __restrict__ pAttn, const float* __restrict__ battn,
            const _Float16* __restrict__ pOut, const float* __restrict__ bout,
            const float* __restrict__ g1, const float* __restrict__ be1,
            const _Float16* __restrict__ pW1, const float* __restrict__ b1,
            const _Float16* __restrict__ pW2, const float* __restrict__ b2,
            const float* __restrict__ g2, const float* __restrict__ be2,
            float* __restrict__ out) {
    __shared__ float sQ[16 * 128];                       // q f32, later pre-LN2 y
    __shared__ alignas(16) _Float16 sB[16 * 128];        // row-major f16 A tile
    __shared__ alignas(16) float sC[16 * 288];           // off+aw; reused: hidden
    __shared__ float sRed[512];                          // LN partials
    __shared__ float sX[16 * 128];                       // x (residual for ffn2)
    _Float16* sH = (_Float16*)sC;                        // row-major hidden 16x512

    const int tid = threadIdx.x, wave = tid >> 5, lane = tid & 31;
    const int vox0 = blockIdx.x << 4;

    // warm L2 for the packed FFN weights (global_prefetch_b8)
    __builtin_prefetch(pW1 + (size_t)tid * 512, 0, 1);
    __builtin_prefetch(pW2 + (size_t)tid * 512, 0, 1);

    // ---- load q tile (16 x 128): each thread owns one 8-float run ----
    {
        int r = tid >> 4, c8 = (tid & 15) * 8;
        const float4* qp = (const float4*)(qg + (size_t)(vox0 + r) * EMBED + c8);
        float4 fa = qp[0], fb = qp[1];
        float4* sq = (float4*)(sQ + r * 128 + c8);
        sq[0] = fa; sq[1] = fb;
        alignas(16) _Float16 t[8] = {
            (_Float16)fa.x, (_Float16)fa.y, (_Float16)fa.z, (_Float16)fa.w,
            (_Float16)fb.x, (_Float16)fb.y, (_Float16)fb.z, (_Float16)fb.w};
        *(v8h*)(sB + r * 128 + c8) = *(const v8h*)t;
    }
    __syncthreads();

    // ---- q projections: 12 off-tiles (192 cols) + 6 attn-tiles (96 cols) ----
    for (int r = 0; r < 3; ++r) {
        int t = wave + r * 8;
        if (t < 18) {
            bool isOff = t < 12;
            int nt = isOff ? t : (t - 12);
            const _Float16* P = isOff ? pOff : pAttn;
            v8f c = {};
#pragma unroll
            for (int ks = 0; ks < 4; ++ks) {
                v16h a = ldsA(sB, lane, 128, ks * 32);
                v16h b = pkB(P, 4, ks, nt, lane);
                c = wmma_f16(a, b, c);
            }
            int n = lane & 15, mo = (lane & 16) ? 8 : 0;
#pragma unroll
            for (int v = 0; v < 8; ++v) {
                int m = v + mo, col = nt * 16 + n;
                if (isOff) sC[m * 288 + col]       = c[v] + boff[col];
                else       sC[m * 288 + 192 + col] = c[v] + battn[col];
            }
        }
    }
    __syncthreads();

    // ---- softmax + deformable sampling: task=(voxel m, head); 2 thr/task ----
    {
        int task = tid >> 1, m = task >> 3, hh = task & 7;
        int dlo = (tid & 1) * 8;
        int voxel = vox0 + m;
        float rx = refpix[(size_t)voxel * 2 + 0];
        float ry = refpix[(size_t)voxel * 2 + 1];
        const float* offp = &sC[m * 288 + hh * 24];
        const float* awp  = &sC[m * 288 + 192 + hh * 12];

        float sm[12];
        float mx = awp[0];
#pragma unroll
        for (int i = 1; i < 12; ++i) mx = fmaxf(mx, awp[i]);
        float den = 0.0f;
#pragma unroll
        for (int i = 0; i < 12; ++i) { sm[i] = expf(awp[i] - mx); den += sm[i]; }
        float rden = 1.0f / den;

        float acc[8];
#pragma unroll
        for (int d = 0; d < 8; ++d) acc[d] = 0.0f;

        for (int l = 0; l < 3; ++l) {
            float wl = (float)c_lw[l], hl = (float)c_lh[l];
            int wi = c_lw[l], st = c_ls[l];
#pragma unroll
            for (int p = 0; p < 4; ++p) {
                float ox = offp[l * 8 + p * 2 + 0];
                float oy = offp[l * 8 + p * 2 + 1];
                float alp = sm[l * 4 + p] * rden;
                float x = (rx + ox / wl) * wl - 0.5f;
                float y = (ry + oy / hl) * hl - 0.5f;
                float x0 = floorf(x), y0 = floorf(y);
#pragma unroll
                for (int dx = 0; dx < 2; ++dx)
#pragma unroll
                    for (int dy = 0; dy < 2; ++dy) {
                        float xi = x0 + (float)dx, yi = y0 + (float)dy;
                        float wgt = (1.0f - fabsf(x - xi)) * (1.0f - fabsf(y - yi));
                        bool valid = (xi >= 0.0f) && (xi <= wl - 1.0f) &&
                                     (yi >= 0.0f) && (yi <= hl - 1.0f);
                        float coef = valid ? wgt * alp : 0.0f;
                        int cx = (int)fminf(fmaxf(xi, 0.0f), wl - 1.0f);
                        int cy = (int)fminf(fmaxf(yi, 0.0f), hl - 1.0f);
                        const float4* g4 = (const float4*)(
                            val + (size_t)(st + cy * wi + cx) * EMBED + hh * DH + dlo);
                        float4 va = g4[0], vb = g4[1];
                        acc[0] += coef * va.x; acc[1] += coef * va.y;
                        acc[2] += coef * va.z; acc[3] += coef * va.w;
                        acc[4] += coef * vb.x; acc[5] += coef * vb.y;
                        acc[6] += coef * vb.z; acc[7] += coef * vb.w;
                    }
            }
        }
        alignas(16) _Float16 t[8];
#pragma unroll
        for (int d = 0; d < 8; ++d) t[d] = (_Float16)acc[d];
        *(v8h*)(sB + m * 128 + hh * DH + dlo) = *(const v8h*)t;
    }
    __syncthreads();

    // ---- out-proj + bias + residual(q) -> sX (pre-LN1) ----
    {
        v8f c = {};
#pragma unroll
        for (int ks = 0; ks < 4; ++ks) {
            v16h a = ldsA(sB, lane, 128, ks * 32);
            v16h b = pkB(pOut, 4, ks, wave, lane);
            c = wmma_f16(a, b, c);
        }
        int n = lane & 15, mo = (lane & 16) ? 8 : 0;
        int col = wave * 16 + n;
        float bb = bout[col];
#pragma unroll
        for (int v = 0; v < 8; ++v) {
            int m = v + mo;
            sX[m * 128 + col] = c[v] + bb + sQ[m * 128 + col];
        }
    }
    __syncthreads();

    // ---- LN1: sX -> sX (f32) and sB (f16, vector store) ----
    {
        int row = tid >> 4, sub = tid & 15;
        float s = 0.0f, ss = 0.0f;
#pragma unroll
        for (int j = 0; j < 8; ++j) {
            float v = sX[row * 128 + sub * 8 + j];
            s += v; ss += v * v;
        }
        sRed[row * 16 + sub] = s;
        sRed[256 + row * 16 + sub] = ss;
        __syncthreads();
        float su = 0.0f, ssu = 0.0f;
#pragma unroll
        for (int j = 0; j < 16; ++j) { su += sRed[row * 16 + j]; ssu += sRed[256 + row * 16 + j]; }
        float mu = su * (1.0f / 128.0f);
        float var = ssu * (1.0f / 128.0f) - mu * mu;
        float rstd = rsqrtf(var + 1e-5f);
        alignas(16) _Float16 t[8];
#pragma unroll
        for (int j = 0; j < 8; ++j) {
            int col = sub * 8 + j;
            float v = (sX[row * 128 + col] - mu) * rstd * g1[col] + be1[col];
            sX[row * 128 + col] = v;
            t[j] = (_Float16)v;
        }
        *(v8h*)(sB + row * 128 + sub * 8) = *(const v8h*)t;
    }
    __syncthreads();

    // ---- FFN1 + exact GELU -> sH (row-major f16, 16x512) ----
    for (int r = 0; r < 4; ++r) {
        int t = wave + r * 8;          // n-tile 0..31
        v8f c = {};
#pragma unroll
        for (int ks = 0; ks < 4; ++ks) {
            v16h a = ldsA(sB, lane, 128, ks * 32);
            v16h b = pkB(pW1, 4, ks, t, lane);
            c = wmma_f16(a, b, c);
        }
        int n = lane & 15, mo = (lane & 16) ? 8 : 0;
        int col = t * 16 + n;
        float bb = b1[col];
#pragma unroll
        for (int v = 0; v < 8; ++v) {
            int m = v + mo;
            float h = c[v] + bb;
            h = 0.5f * h * (1.0f + erff(h * 0.70710678118654752f));
            sH[m * 512 + col] = (_Float16)h;
        }
    }
    __syncthreads();

    // ---- FFN2 + bias + residual(x) -> sQ (pre-LN2) ----
    {
        v8f c = {};
#pragma unroll
        for (int ks = 0; ks < 16; ++ks) {
            v16h a = ldsA(sH, lane, 512, ks * 32);
            v16h b = pkB(pW2, 16, ks, wave, lane);
            c = wmma_f16(a, b, c);
        }
        int n = lane & 15, mo = (lane & 16) ? 8 : 0;
        int col = wave * 16 + n;
        float bb = b2[col];
#pragma unroll
        for (int v = 0; v < 8; ++v) {
            int m = v + mo;
            sQ[m * 128 + col] = c[v] + bb + sX[m * 128 + col];
        }
    }
    __syncthreads();

    // ---- LN2 + masked select -> out (float4 stores) ----
    {
        int row = tid >> 4, sub = tid & 15;
        float s = 0.0f, ss = 0.0f;
#pragma unroll
        for (int j = 0; j < 8; ++j) {
            float v = sQ[row * 128 + sub * 8 + j];
            s += v; ss += v * v;
        }
        sRed[row * 16 + sub] = s;
        sRed[256 + row * 16 + sub] = ss;
        __syncthreads();
        float su = 0.0f, ssu = 0.0f;
#pragma unroll
        for (int j = 0; j < 16; ++j) { su += sRed[row * 16 + j]; ssu += sRed[256 + row * 16 + j]; }
        float mu = su * (1.0f / 128.0f);
        float var = ssu * (1.0f / 128.0f) - mu * mu;
        float rstd = rsqrtf(var + 1e-5f);
        int voxel = vox0 + row;
        bool keep = mask[voxel] != 0;
        float vv[8];
#pragma unroll
        for (int j = 0; j < 8; ++j) {
            int col = sub * 8 + j;
            vv[j] = (sQ[row * 128 + col] - mu) * rstd * g2[col] + be2[col];
        }
        float4 r0, r1;
        if (keep) {
            r0 = make_float4(vv[0], vv[1], vv[2], vv[3]);
            r1 = make_float4(vv[4], vv[5], vv[6], vv[7]);
        } else {
            const float4* qp = (const float4*)(qg + (size_t)voxel * EMBED + sub * 8);
            r0 = qp[0]; r1 = qp[1];
        }
        float4* op = (float4*)(out + (size_t)voxel * EMBED + sub * 8);
        op[0] = r0; op[1] = r1;
    }
}

// ---------------------------------------------------------------------------
// Host launcher
// ---------------------------------------------------------------------------
extern "C" void kernel_launch(void* const* d_in, const int* in_sizes, int n_in,
                              void* d_out, int out_size, void* d_ws, size_t ws_size,
                              hipStream_t stream) {
    (void)in_sizes; (void)n_in; (void)out_size; (void)ws_size;
    const float* scene  = (const float*)d_in[0];
    const float* f0     = (const float*)d_in[1];
    const float* f1     = (const float*)d_in[2];
    const float* f2     = (const float*)d_in[3];
    const float* depth  = (const float*)d_in[4];
    const float* Km     = (const float*)d_in[5];
    const float* Em     = (const float*)d_in[6];
    const float* origin = (const float*)d_in[7];
    const float* refpix = (const float*)d_in[8];
    const float* Woff   = (const float*)d_in[9];
    const float* boff   = (const float*)d_in[10];
    const float* Wattn  = (const float*)d_in[11];
    const float* battn  = (const float*)d_in[12];
    const float* Wval   = (const float*)d_in[13];
    const float* bval   = (const float*)d_in[14];
    const float* Wout   = (const float*)d_in[15];
    const float* bout   = (const float*)d_in[16];
    const float* g1     = (const float*)d_in[17];
    const float* be1    = (const float*)d_in[18];
    const float* W1     = (const float*)d_in[19];
    const float* b1     = (const float*)d_in[20];
    const float* W2     = (const float*)d_in[21];
    const float* b2     = (const float*)d_in[22];
    const float* g2     = (const float*)d_in[23];
    const float* be2    = (const float*)d_in[24];
    float* out = (float*)d_out;

    // workspace layout (float units):
    //   [0,64)            invK (9) + invE (16)
    //   [64, 64+NVOXT)    occupancy mask (int32)
    //   value:            NVF x 128 f32 sampled-value table
    //   packed f16:       Woff,Wattn,Wval,Wout,W1,W2 in WMMA-B layout, feat-A
    float* ws    = (float*)d_ws;
    float* invKE = ws;
    int*   mask  = (int*)(ws + 64);
    float* value = ws + 64 + NVOXT;
    _Float16* pk   = (_Float16*)(ws + 64 + NVOXT + (size_t)NVF * EMBED);
    _Float16* pOff  = pk;                 // 4*12*32*16  = 24576
    _Float16* pAttn = pOff  + 24576;      // 4*6*32*16   = 12288
    _Float16* pVal  = pAttn + 12288;      // 4*8*32*16   = 16384
    _Float16* pOut  = pVal  + 16384;      // 4*8*32*16   = 16384
    _Float16* pW1   = pOut  + 16384;      // 4*32*32*16  = 65536
    _Float16* pW2   = pW1   + 65536;      // 16*8*32*16  = 65536
    _Float16* pA    = pW2   + 65536;      // 598*4*32*16 = 1224704

    k_setup<<<1, 32, 0, stream>>>(Km, Em, invKE);
    k_zero<<<(NVOXT + 255) / 256, 256, 0, stream>>>(mask);
    k_project<<<(IMGH * IMGW + 255) / 256, 256, 0, stream>>>(depth, invKE, origin, mask);

    k_packW<<<(4 * 12 * 32 + 255) / 256, 256, 0, stream>>>(Woff, 4, 192, pOff);
    k_packW<<<(4 * 6  * 32 + 255) / 256, 256, 0, stream>>>(Wattn, 4, 96, pAttn);
    k_packW<<<(4 * 8  * 32 + 255) / 256, 256, 0, stream>>>(Wval, 4, 128, pVal);
    k_packW<<<(4 * 8  * 32 + 255) / 256, 256, 0, stream>>>(Wout, 4, 128, pOut);
    k_packW<<<(4 * 32 * 32 + 255) / 256, 256, 0, stream>>>(W1, 4, 512, pW1);
    k_packW<<<(16 * 8 * 32 + 255) / 256, 256, 0, stream>>>(W2, 16, 128, pW2);
    k_packA<<<(MT_VAL * 4 * 32 + 255) / 256, 256, 0, stream>>>(f0, f1, f2, pA);

    k_value<<<MT_VAL, 256, 0, stream>>>(pA, pVal, bval, value);
    k_mega<<<NVOXT / 16, 256, 0, stream>>>(scene, refpix, value, mask,
                                           pOff, boff, pAttn, battn,
                                           pOut, bout, g1, be1,
                                           pW1, b1, pW2, b2, g2, be2, out);
}
